// SpectralPatchEmbeddingPerBand_70257075028350
// MI455X (gfx1250) — matrix operand order repaired and checked by reading, not compiled, not run on hardware
//
#include <hip/hip_runtime.h>
#include <hip/hip_bf16.h>

// ---------------------------------------------------------------------------
// SpectralPatchEmbeddingPerBand for MI455X (gfx1250, wave32, WMMA).
//
// B=512, C=13, H=64, P=8, D_BAND=64, D=832, N=64 patches, K=P*P=64.
// Fused: patch-projection (per band) -> y kept in LDS (bf16) -> fusion GEMM.
// Math path: bf16 operands, f32 accumulation via v_wmma_f32_16x16x32_bf16.
//
// v2: 2 workgroups per batch image (32 patch rows each), 256 threads = 8
// waves, 54.3 KB LDS (under the 64 KB dynamic-LDS launch cap, ~5 WG/WGP).
// ---------------------------------------------------------------------------

typedef __bf16 bf16_t;
typedef __attribute__((ext_vector_type(16))) __bf16 v16bf;
typedef __attribute__((ext_vector_type(8)))  __bf16 v8bf;
typedef __attribute__((ext_vector_type(8)))  float  v8f;

#define NB   512
#define NC   13
#define NPAT 64          // patches per image (8x8)
#define KK   64          // P*P
#define DB   64          // D_BAND
#define DD   832         // D = NC*DB
#define YST  848         // LDS row stride in bf16 elements (832 + 16 pad)
#define YROWS 32         // patch rows staged per workgroup

// ---------------------------------------------------------------------------
// Prep 1: patchify x (f32) -> xp (bf16), layout xp[b][c][n][k], k contiguous.
// One thread converts one 8-element (16B) patch row chunk.
// total = B*C*64*8 = 3,407,872 chunks.
// ---------------------------------------------------------------------------
__global__ void __launch_bounds__(256) prep_x_kernel(
    const float* __restrict__ x, bf16_t* __restrict__ xp, int total)
{
    int o = blockIdx.x * blockDim.x + threadIdx.x;
    if (o >= total) return;
    int py = o & 7;
    int n  = (o >> 3) & 63;
    int bc = o >> 9;                 // b*13 + c
    int ph = n >> 3, pw = n & 7;
    const float* s = x  + ((size_t)bc * 64 + (size_t)(ph * 8 + py)) * 64 + pw * 8;
    bf16_t*      d = xp + ((size_t)bc * 64 + (size_t)n) * 64 + py * 8;
    v8bf v;
#pragma unroll
    for (int i = 0; i < 8; ++i) v[i] = (bf16_t)s[i];
    *(v8bf*)d = v;
}

// ---------------------------------------------------------------------------
// Prep 2: convert conv_w (C,DB,P,P) and fusion_w (D,D) f32 -> bf16 (same layout).
// ---------------------------------------------------------------------------
__global__ void __launch_bounds__(256) prep_w_kernel(
    const float* __restrict__ conv_w, const float* __restrict__ fusion_w,
    bf16_t* __restrict__ wkb, bf16_t* __restrict__ wfb)
{
    int i = blockIdx.x * blockDim.x + threadIdx.x;
    const int NK = NC * DB * KK;     // 53248
    const int NF = DD * DD;          // 692224
    if (i < NK) wkb[i] = (bf16_t)conv_w[i];
    if (i < NF) wfb[i] = (bf16_t)fusion_w[i];
}

// ---------------------------------------------------------------------------
// Main fused kernel: 2 blocks per b (32 patch rows each), 256 threads = 8
// waves (wave32). Wave (r = w&1, g = w>>1):
//   Stage A: for band c in 0..12 compute y tile rows [rowOff+16r, +16) cols
//            [c*64+16g, +16) via 2x wmma (K=64), add conv_b, store bf16 to LDS.
//   Stage B: accumulate out tiles (rows rowOff+16r, cols (g+4j)*16, j=0..12)
//            over K=832 in 26 steps of 32; A from LDS, B (fusion_w bf16) from
//            global (WGP$/L2-resident, immediate-offset b128 loads).
// ---------------------------------------------------------------------------
__global__ void __launch_bounds__(256) spectral_fused_kernel(
    const bf16_t* __restrict__ xp,        // [B][C][64][64] bf16
    const bf16_t* __restrict__ wkb,       // [C][64][64]    bf16
    const bf16_t* __restrict__ wfb,       // [832][832]     bf16 (row = out e)
    const float*  __restrict__ conv_b,    // [C*DB] = [832]
    const float*  __restrict__ fusion_b,  // [832]
    float*        __restrict__ out)       // [B][64][832]
{
    extern __shared__ char smem[];
    bf16_t* ylds = (bf16_t*)smem;         // [YROWS][YST]

    const int b      = blockIdx.x >> 1;
    const int rowOff = (blockIdx.x & 1) * YROWS;  // 0 or 32 (patch-row offset)
    const int t  = threadIdx.x & 31;
    const int w  = threadIdx.x >> 5;
    const int r  = w & 1;                 // row-block within half-image, 0..1
    const int g  = w >> 1;                // col-group 0..3
    const int tn = t & 15;                // lane's N index (and A-row index)
    const int th = t >> 4;                // lane half 0/1

    // ---------------- Stage A: per-band projection into LDS ----------------
    {
        const int mA = rowOff + r * 16 + tn;                // global patch index
        const size_t xb = (size_t)b * NC * NPAT * KK;       // x base for this b
        const int d0 = g * 16;                              // within-band col offset

        for (int c = 0; c < NC; ++c) {
            v8f acc = {};
#pragma unroll
            for (int half = 0; half < 2; ++half) {
                // A fragment: 16x32 bf16. lanes<16: K 0..15; lanes>=16: K 16..31.
                const int kA = half * 32 + th * 16;
                const bf16_t* ap = xp + xb + ((size_t)c * NPAT + mA) * KK + kA;
                v8bf lo = *(const v8bf*)ap;
                v8bf hi = *(const v8bf*)(ap + 8);
                v16bf a = __builtin_shufflevector(lo, hi,
                    0, 1, 2, 3, 4, 5, 6, 7, 8, 9, 10, 11, 12, 13, 14, 15);
                // B fragment: 32x16 bf16. B[k][n] = Wc[d0+n][k]; lanes<16: K 0..15.
                const int kB = half * 32 + th * 16;
                const bf16_t* bp = wkb + ((size_t)c * DB + d0 + tn) * KK + kB;
                v16bf bb = *(const v16bf*)bp;
                acc = __builtin_amdgcn_wmma_f32_16x16x32_bf16(
                    false, a, false, bb, (short)0, acc, false, false);
            }
            // Store tile to LDS (+conv_b). Lane holds column d, rows mBase..+7.
            const int d  = c * DB + d0 + tn;
            const float cb = conv_b[d];
            const int mBase = r * 16 + th * 8;                // local LDS row
#pragma unroll
            for (int i = 0; i < 8; ++i)
                ylds[(mBase + i) * YST + d] = (bf16_t)(acc[i] + cb);
        }
    }

    __syncthreads();

    // ---------------- Stage B: fusion GEMM out = y @ Wf^T + bias -----------
    v8f acc[13];
#pragma unroll
    for (int j = 0; j < 13; ++j) { v8f z = {}; acc[j] = z; }

    const int mB = r * 16 + tn;                 // local LDS row for lane's A row
    const bf16_t* arow = ylds + (size_t)mB * YST;
    const int kaOff = th * 16;
    const int kbOff = th * 16;
    const int eCol  = g * 16 + tn;              // lane's output column base (tile j adds 64j)

    for (int ks = 0; ks < 26; ++ks) {
        const int k0 = ks * 32;
        v8bf lo = *(const v8bf*)(arow + k0 + kaOff);
        v8bf hi = *(const v8bf*)(arow + k0 + kaOff + 8);
        v16bf a = __builtin_shufflevector(lo, hi,
            0, 1, 2, 3, 4, 5, 6, 7, 8, 9, 10, 11, 12, 13, 14, 15);
#pragma unroll
        for (int j = 0; j < 13; ++j) {
            const bf16_t* bp = wfb + (size_t)(eCol + 64 * j) * DD + k0 + kbOff;
            v16bf bb = *(const v16bf*)bp;
            acc[j] = __builtin_amdgcn_wmma_f32_16x16x32_bf16(
                false, a, false, bb, (short)0, acc[j], false, false);
        }
    }

    // ---------------- Epilogue: bias + f32 store ---------------------------
    const int mBase = rowOff + r * 16 + th * 8;               // global out row
#pragma unroll
    for (int j = 0; j < 13; ++j) {
        const int e = eCol + 64 * j;
        const float bias = fusion_b[e];
        float* op = out + ((size_t)b * NPAT + mBase) * DD + e;
#pragma unroll
        for (int i = 0; i < 8; ++i)
            op[(size_t)i * DD] = acc[j][i] + bias;
    }
}

// ---------------------------------------------------------------------------
// Launch
// ---------------------------------------------------------------------------
extern "C" void kernel_launch(void* const* d_in, const int* in_sizes, int n_in,
                              void* d_out, int out_size, void* d_ws, size_t ws_size,
                              hipStream_t stream)
{
    const float* x        = (const float*)d_in[0];
    const float* conv_w   = (const float*)d_in[1];
    const float* conv_b   = (const float*)d_in[2];
    const float* fusion_w = (const float*)d_in[3];
    const float* fusion_b = (const float*)d_in[4];
    float* out = (float*)d_out;

    // Workspace layout (bf16): xp | wkb | wfb   (~53.4 MiB total)
    bf16_t* xp  = (bf16_t*)d_ws;                         // 27,262,976 elems
    bf16_t* wkb = xp  + (size_t)NB * NC * NPAT * KK;     // 53,248 elems
    bf16_t* wfb = wkb + (size_t)NC * DB * KK;            // 692,224 elems

    // Prep: patchify + bf16 conversion.
    const int xchunks = NB * NC * NPAT * 8;              // 3,407,872
    prep_x_kernel<<<xchunks / 256, 256, 0, stream>>>(x, xp, xchunks);
    prep_w_kernel<<<(DD * DD) / 256, 256, 0, stream>>>(conv_w, fusion_w, wkb, wfb);

    // Fused main kernel: 2 blocks per batch image, 8 waves, 54.3 KB LDS.
    const size_t smem_bytes = (size_t)YROWS * YST * sizeof(bf16_t);
    spectral_fused_kernel<<<NB * 2, 256, smem_bytes, stream>>>(
        xp, wkb, wfb, conv_b, fusion_b, out);
}
